// MultiHeadAttention_3393024164528
// MI455X (gfx1250) — compile-verified, matrix operand order checked by reading, not run on hardware
//
#include <hip/hip_runtime.h>
#include <hip/hip_bf16.h>

// ---------------------------------------------------------------------------
// MHA forward for MI455X (gfx1250): bf16 WMMA everywhere, flash-attention
// fusion (no (B,H,S,S) score tensor in HBM), async-to-LDS double-buffered
// K/V streaming.
//   B=2, S=4096, D=768, H=12, dk=64
// ---------------------------------------------------------------------------

#define SEQ    4096
#define DMODEL 768
#define HEADS  12
#define DKH    64
#define BATCH  2
#define NJ     (SEQ / 64)      // key blocks per head

#if defined(__AMDGCN__) && __has_builtin(__builtin_amdgcn_global_load_async_to_lds_b128)
#define HAVE_ASYNC 1
#else
#define HAVE_ASYNC 0
#endif

typedef __attribute__((ext_vector_type(16))) __bf16 v16bf;
typedef __attribute__((ext_vector_type(8)))  float  v8f;

union AFrag { v16bf v; unsigned u[8]; };

// 16-bit A/B fragment K-pattern (ISA 7.12.2): dword i of lane-half `lh`
// holds K = pairK..pairK+1.
__device__ __forceinline__ int pairK(int i, int lh) {
  return ((i < 4) ? (2 * i) : (2 * i + 8)) + 8 * lh;
}

__device__ __forceinline__ unsigned short f2bf(float f) {
  unsigned u = __builtin_bit_cast(unsigned, f);
  unsigned r = u + 0x7FFFu + ((u >> 16) & 1u);   // round-to-nearest-even
  return (unsigned short)(r >> 16);
}

__device__ __forceinline__ v8f wmma_bf16(const AFrag& a, const AFrag& b, v8f c) {
  // D(16x16,f32) = A(16x32,bf16) * B(32x16,bf16) + C
  return __builtin_amdgcn_wmma_f32_16x16x32_bf16(false, a.v, false, b.v,
                                                 (short)0, c, false, false);
}

#if HAVE_ASYNC
// Builtin signature (from hipcc diagnostic): params are pointers to
// vector_size(16) int, global (AS1) source and LDS (AS3) destination.
typedef int v4i_vs __attribute__((vector_size(16)));
typedef __attribute__((address_space(1))) v4i_vs gbl_v4i;
typedef __attribute__((address_space(3))) v4i_vs lds_v4i;

// one lane moves 16 B HBM -> LDS via the async engine (ASYNCcnt-tracked)
__device__ __forceinline__ void async_cp16(const unsigned short* g,
                                           unsigned short* l) {
  __builtin_amdgcn_global_load_async_to_lds_b128(
      (gbl_v4i*)g, (lds_v4i*)l, 0, 0);
}

__device__ __forceinline__ void wait_async_le(int n) {
#if __has_builtin(__builtin_amdgcn_s_wait_asynccnt)
  if (n == 0) __builtin_amdgcn_s_wait_asynccnt(0);
  else        __builtin_amdgcn_s_wait_asynccnt(4);
#else
  if (n == 0) asm volatile("s_wait_asynccnt 0x0" ::: "memory");
  else        asm volatile("s_wait_asynccnt 0x4" ::: "memory");
#endif
}
#endif

// ---------------------------------------------------------------------------
// Kernel 1: q/k/v projection.  q,k -> (B,H,S,64) bf16;  v -> (B,H,64,S) bf16
// (V transposed so the flash kernel's PV tiles are straight async copies).
// block = 256 thr (8 waves) -> 128x64 output tile, K-steps of 32 via LDS.
// ---------------------------------------------------------------------------
__global__ __launch_bounds__(256) void proj_qkv_kernel(
    const float* __restrict__ Q, const float* __restrict__ K,
    const float* __restrict__ V,
    const float* __restrict__ Wq, const float* __restrict__ Wk,
    const float* __restrict__ Wv,
    const float* __restrict__ bq, const float* __restrict__ bk,
    const float* __restrict__ bv,
    unsigned short* __restrict__ qh, unsigned short* __restrict__ kh,
    unsigned short* __restrict__ vh)
{
  __shared__ __attribute__((aligned(16))) unsigned short As[128][34];
  __shared__ __attribute__((aligned(16))) unsigned short Bs[64][34];

  const int tid  = threadIdx.x;
  const int lane = tid & 31;
  const int w    = tid >> 5;
  const int l16  = lane & 15;
  const int lh   = lane >> 4;

  const float* X; const float* W; const float* bias; unsigned short* dst;
  if (blockIdx.z == 0)      { X = Q; W = Wq; bias = bq; dst = qh; }
  else if (blockIdx.z == 1) { X = K; W = Wk; bias = bk; dst = kh; }
  else                      { X = V; W = Wv; bias = bv; dst = vh; }
  const bool vT = (blockIdx.z == 2);

  const int m0 = blockIdx.y * 128;
  const int n0 = blockIdx.x * 64;

  v8f acc[4] = {};

  const int arow = tid >> 1;          // 0..127
  const int acol = (tid & 1) * 16;    // 0 / 16
  const int brow = tid >> 2;          // 0..63
  const int bcol = (tid & 3) * 8;     // 0,8,16,24

  for (int k0 = 0; k0 < DMODEL; k0 += 32) {
    __syncthreads();
    { // stage X tile, f32 -> bf16
      const float4* src = (const float4*)(X + (size_t)(m0 + arow) * DMODEL + k0 + acol);
      #pragma unroll
      for (int i = 0; i < 4; ++i) {
        float4 f = src[i];
        int c = acol + 4 * i;
        As[arow][c + 0] = f2bf(f.x); As[arow][c + 1] = f2bf(f.y);
        As[arow][c + 2] = f2bf(f.z); As[arow][c + 3] = f2bf(f.w);
      }
    }
    { // stage W tile (row n = output col, contiguous K), f32 -> bf16
      const float4* src = (const float4*)(W + (size_t)(n0 + brow) * DMODEL + k0 + bcol);
      #pragma unroll
      for (int i = 0; i < 2; ++i) {
        float4 f = src[i];
        int c = bcol + 4 * i;
        Bs[brow][c + 0] = f2bf(f.x); Bs[brow][c + 1] = f2bf(f.y);
        Bs[brow][c + 2] = f2bf(f.z); Bs[brow][c + 3] = f2bf(f.w);
      }
    }
    __syncthreads();

    AFrag a;
    #pragma unroll
    for (int i = 0; i < 8; ++i)
      a.u[i] = *(const unsigned*)&As[16 * w + l16][pairK(i, lh)];

    #pragma unroll
    for (int nt = 0; nt < 4; ++nt) {
      AFrag bf;
      #pragma unroll
      for (int i = 0; i < 8; ++i)
        bf.u[i] = *(const unsigned*)&Bs[16 * nt + l16][pairK(i, lh)];
      acc[nt] = wmma_bf16(a, bf, acc[nt]);
    }
  }

  // epilogue: bias + bf16 store
  #pragma unroll
  for (int nt = 0; nt < 4; ++nt) {
    #pragma unroll
    for (int r = 0; r < 8; ++r) {
      int m = m0 + 16 * w + r + 8 * lh;      // global row (b*S+s)
      int c = n0 + 16 * nt + l16;            // global col (h*64+d)
      float val = acc[nt][r] + bias[c];
      int bidx = m >> 12;
      int s    = m & (SEQ - 1);
      int hh   = c >> 6;
      int d    = c & 63;
      size_t idx = vT
        ? ((((size_t)bidx * HEADS + hh) * DKH + d) * SEQ + s)   // (B,H,dk,S)
        : ((((size_t)bidx * HEADS + hh) * SEQ + s) * DKH + d);  // (B,H,S,dk)
      dst[idx] = f2bf(val);
    }
  }
}

// ---------------------------------------------------------------------------
// Kernel 2: flash attention.  One (b, h, 128 q-rows) per block; wave owns 16
// q rows.  64-key blocks double-buffered through LDS via async-to-LDS copies
// (ASYNCcnt pipelined against WMMA compute); online softmax; P·V via WMMA.
// ---------------------------------------------------------------------------
__global__ __launch_bounds__(256) void flash_attn_kernel(
    const unsigned short* __restrict__ qh,
    const unsigned short* __restrict__ kh,
    const unsigned short* __restrict__ vh,     // (B,H,dk,S)
    unsigned short* __restrict__ combined)
{
  // 72-halfword rows: 144 B = 16 B-aligned async chunks, 36-bank stride
  // (conflict-free for 16-row fragment reads).
  __shared__ __attribute__((aligned(16))) unsigned short Ks[2][64][72]; // [key][dk]
  __shared__ __attribute__((aligned(16))) unsigned short Vt[2][64][72]; // [dk][key]
  __shared__ __attribute__((aligned(16))) unsigned short Ps[8][16][66]; // per-wave P

  const int tid  = threadIdx.x;
  const int lane = tid & 31;
  const int w    = tid >> 5;
  const int l16  = lane & 15;
  const int lh   = lane >> 4;

  const int b  = blockIdx.z;
  const int h  = blockIdx.y;
  const int s0 = blockIdx.x * 128;

  const size_t headBase = ((size_t)b * HEADS + h) * SEQ * DKH;
  const unsigned short* kBaseG = kh + headBase;   // [key][dk]
  const unsigned short* vBaseG = vh + headBase;   // [dk][key]

  // q A-fragments for this wave's 16 rows (K = 0..63 -> two frags)
  AFrag qa[2];
  {
    const unsigned short* qrow = qh + headBase + (size_t)(s0 + 16 * w + l16) * DKH;
    #pragma unroll
    for (int kk = 0; kk < 2; ++kk)
      #pragma unroll
      for (int i = 0; i < 8; ++i)
        qa[kk].u[i] = *(const unsigned*)(qrow + 32 * kk + pairK(i, lh));
  }

  v8f ao[4] = {};                 // output accum: 16 rows x 64 dk
  float mi[8], li[8];
  #pragma unroll
  for (int r = 0; r < 8; ++r) { mi[r] = -3.0e38f; li[r] = 0.f; }

#if HAVE_ASYNC
  // 512 x 16B chunks per (K,V) tile; 256 threads issue 2 chunks each.
  auto issueTile = [&](int buf, int j) {
    #pragma unroll
    for (int i = 0; i < 2; ++i) {
      int q   = tid + 256 * i;   // 0..511
      int row = q >> 3;          // 0..63
      int c8  = (q & 7) * 8;     // bf16 offset of 16B chunk within row
      async_cp16(kBaseG + ((size_t)(j * 64 + row)) * DKH + c8, &Ks[buf][row][c8]);
      async_cp16(vBaseG + (size_t)row * SEQ + j * 64 + c8,     &Vt[buf][row][c8]);
    }
  };
  issueTile(0, 0);
#endif

  for (int j = 0; j < NJ; ++j) {
    const int cur = j & 1;
    __syncthreads();                       // everyone done with buffer cur^1

#if HAVE_ASYNC
    if (j + 1 < NJ) { issueTile(cur ^ 1, j + 1); wait_async_le(4); }
    else            { wait_async_le(0); }
#else
    { // synchronous fallback staging (both tiles are straight copies now)
      const int ldrow = tid >> 2;          // 0..63
      const int ldcol = (tid & 3) * 16;    // 0,16,32,48
      {
        const uint4* src = (const uint4*)(kBaseG + (size_t)(j * 64 + ldrow) * DKH + ldcol);
        uint4 f0 = src[0], f1 = src[1];
        unsigned* drow = (unsigned*)&Ks[cur][ldrow][ldcol];
        drow[0] = f0.x; drow[1] = f0.y; drow[2] = f0.z; drow[3] = f0.w;
        drow[4] = f1.x; drow[5] = f1.y; drow[6] = f1.z; drow[7] = f1.w;
      }
      {
        const uint4* src = (const uint4*)(vBaseG + (size_t)ldrow * SEQ + j * 64 + ldcol);
        uint4 f0 = src[0], f1 = src[1];
        unsigned* drow = (unsigned*)&Vt[cur][ldrow][ldcol];
        drow[0] = f0.x; drow[1] = f0.y; drow[2] = f0.z; drow[3] = f0.w;
        drow[4] = f1.x; drow[5] = f1.y; drow[6] = f1.z; drow[7] = f1.w;
      }
    }
#endif
    __syncthreads();                       // tile cur fully in LDS

    // S = q · kᵀ  (16 x 64 scores per wave)
    v8f sc[4] = {};
    #pragma unroll
    for (int kk = 0; kk < 2; ++kk) {
      #pragma unroll
      for (int nt = 0; nt < 4; ++nt) {
        AFrag kf;
        #pragma unroll
        for (int i = 0; i < 8; ++i)
          kf.u[i] = *(const unsigned*)&Ks[cur][16 * nt + l16][32 * kk + pairK(i, lh)];
        sc[nt] = wmma_bf16(qa[kk], kf, sc[nt]);
      }
    }

    // online softmax (rows live across 16-lane halves; wave32 shfl_xor)
    float alpha[8];
    #pragma unroll
    for (int r = 0; r < 8; ++r) {
      float mloc = -3.0e38f;
      #pragma unroll
      for (int nt = 0; nt < 4; ++nt) {
        float s = sc[nt][r] * 0.125f;          // 1/sqrt(dk)
        sc[nt][r] = s;
        mloc = fmaxf(mloc, s);
      }
      #pragma unroll
      for (int off = 8; off >= 1; off >>= 1)
        mloc = fmaxf(mloc, __shfl_xor(mloc, off));
      float mnew = fmaxf(mi[r], mloc);
      float a = __expf(mi[r] - mnew);
      mi[r] = mnew;
      alpha[r] = a;
      float rs = 0.f;
      #pragma unroll
      for (int nt = 0; nt < 4; ++nt) {
        float p = __expf(sc[nt][r] - mnew);
        sc[nt][r] = p;
        rs += p;
      }
      #pragma unroll
      for (int off = 8; off >= 1; off >>= 1)
        rs += __shfl_xor(rs, off);
      li[r] = li[r] * a + rs;
    }

    #pragma unroll
    for (int dt = 0; dt < 4; ++dt)
      #pragma unroll
      for (int r = 0; r < 8; ++r)
        ao[dt][r] *= alpha[r];

    // P (C-layout) -> LDS -> A-layout fragments, bf16
    #pragma unroll
    for (int nt = 0; nt < 4; ++nt)
      #pragma unroll
      for (int r = 0; r < 8; ++r)
        Ps[w][r + 8 * lh][16 * nt + l16] = f2bf(sc[nt][r]);

    // in-wave cross-lane LDS dependency: drain DScnt before re-reading
    __builtin_amdgcn_wave_barrier();
    asm volatile("s_wait_dscnt 0" ::: "memory");
    __builtin_amdgcn_wave_barrier();

    // O += P · V
    #pragma unroll
    for (int kk = 0; kk < 2; ++kk) {
      AFrag pa;
      #pragma unroll
      for (int i = 0; i < 8; ++i)
        pa.u[i] = *(const unsigned*)&Ps[w][l16][32 * kk + pairK(i, lh)];
      #pragma unroll
      for (int dt = 0; dt < 4; ++dt) {
        AFrag vf;
        #pragma unroll
        for (int i = 0; i < 8; ++i)
          vf.u[i] = *(const unsigned*)&Vt[cur][16 * dt + l16][32 * kk + pairK(i, lh)];
        ao[dt] = wmma_bf16(pa, vf, ao[dt]);
      }
    }
  }

  // finalize: O / l, store combined (B,S,D) bf16 with heads re-interleaved
  float inv[8];
  #pragma unroll
  for (int r = 0; r < 8; ++r)
    inv[r] = (li[r] > 0.f) ? (1.f / li[r]) : 0.f;

  #pragma unroll
  for (int dt = 0; dt < 4; ++dt)
    #pragma unroll
    for (int r = 0; r < 8; ++r) {
      int srow = s0 + 16 * w + r + 8 * lh;
      int c    = h * DKH + 16 * dt + l16;
      combined[((size_t)b * SEQ + srow) * DMODEL + c] = f2bf(ao[dt][r] * inv[r]);
    }
}

// ---------------------------------------------------------------------------
// Kernel 3: output projection.  out = combined(bf16) · Woᵀ + bo  (f32 out)
// ---------------------------------------------------------------------------
__global__ __launch_bounds__(256) void out_proj_kernel(
    const unsigned short* __restrict__ Xc,
    const float* __restrict__ Wo, const float* __restrict__ bo,
    float* __restrict__ out)
{
  __shared__ __attribute__((aligned(16))) unsigned short As[128][34];
  __shared__ __attribute__((aligned(16))) unsigned short Bs[64][34];

  const int tid  = threadIdx.x;
  const int lane = tid & 31;
  const int w    = tid >> 5;
  const int l16  = lane & 15;
  const int lh   = lane >> 4;

  const int m0 = blockIdx.y * 128;
  const int n0 = blockIdx.x * 64;

  v8f acc[4] = {};

  const int arow = tid >> 1;
  const int acol = (tid & 1) * 16;
  const int brow = tid >> 2;
  const int bcol = (tid & 3) * 8;

  for (int k0 = 0; k0 < DMODEL; k0 += 32) {
    __syncthreads();
    { // stage A tile (already bf16): raw copy
      const uint4* src = (const uint4*)(Xc + (size_t)(m0 + arow) * DMODEL + k0 + acol);
      uint4 f0 = src[0], f1 = src[1];
      unsigned* drow = (unsigned*)&As[arow][acol];
      drow[0] = f0.x; drow[1] = f0.y; drow[2] = f0.z; drow[3] = f0.w;
      drow[4] = f1.x; drow[5] = f1.y; drow[6] = f1.z; drow[7] = f1.w;
    }
    { // stage Wo tile, f32 -> bf16
      const float4* src = (const float4*)(Wo + (size_t)(n0 + brow) * DMODEL + k0 + bcol);
      #pragma unroll
      for (int i = 0; i < 2; ++i) {
        float4 f = src[i];
        int c = bcol + 4 * i;
        Bs[brow][c + 0] = f2bf(f.x); Bs[brow][c + 1] = f2bf(f.y);
        Bs[brow][c + 2] = f2bf(f.z); Bs[brow][c + 3] = f2bf(f.w);
      }
    }
    __syncthreads();

    AFrag a;
    #pragma unroll
    for (int i = 0; i < 8; ++i)
      a.u[i] = *(const unsigned*)&As[16 * w + l16][pairK(i, lh)];
    #pragma unroll
    for (int nt = 0; nt < 4; ++nt) {
      AFrag bf;
      #pragma unroll
      for (int i = 0; i < 8; ++i)
        bf.u[i] = *(const unsigned*)&Bs[16 * nt + l16][pairK(i, lh)];
      acc[nt] = wmma_bf16(a, bf, acc[nt]);
    }
  }

  #pragma unroll
  for (int nt = 0; nt < 4; ++nt)
    #pragma unroll
    for (int r = 0; r < 8; ++r) {
      int m = m0 + 16 * w + r + 8 * lh;
      int c = n0 + 16 * nt + l16;
      out[(size_t)m * DMODEL + c] = acc[nt][r] + bo[c];
    }
}

// ---------------------------------------------------------------------------
extern "C" void kernel_launch(void* const* d_in, const int* in_sizes, int n_in,
                              void* d_out, int out_size, void* d_ws, size_t ws_size,
                              hipStream_t stream) {
  (void)in_sizes; (void)n_in; (void)out_size; (void)ws_size;

  const float* Q  = (const float*)d_in[0];
  const float* K  = (const float*)d_in[1];
  const float* V  = (const float*)d_in[2];
  const float* Wq = (const float*)d_in[3];
  const float* bq = (const float*)d_in[4];
  const float* Wk = (const float*)d_in[5];
  const float* bk = (const float*)d_in[6];
  const float* Wv = (const float*)d_in[7];
  const float* bv = (const float*)d_in[8];
  const float* Wo = (const float*)d_in[9];
  const float* bo = (const float*)d_in[10];
  float* out = (float*)d_out;

  // workspace: q,k head-major + v dk-major + combined, all bf16 (50.3 MB)
  const size_t nHead = (size_t)BATCH * HEADS * SEQ * DKH;
  unsigned short* qh   = (unsigned short*)d_ws;
  unsigned short* kh   = qh + nHead;
  unsigned short* vh   = kh + nHead;
  unsigned short* comb = vh + nHead;

  dim3 blk(256);
  hipLaunchKernelGGL(proj_qkv_kernel,
                     dim3(DMODEL / 64, (BATCH * SEQ) / 128, 3), blk, 0, stream,
                     Q, K, V, Wq, Wk, Wv, bq, bk, bv, qh, kh, vh);
  hipLaunchKernelGGL(flash_attn_kernel,
                     dim3(SEQ / 128, HEADS, BATCH), blk, 0, stream,
                     qh, kh, vh, comb);
  hipLaunchKernelGGL(out_proj_kernel,
                     dim3(DMODEL / 64, (BATCH * SEQ) / 128, 1), blk, 0, stream,
                     comb, Wo, bo, out);
}
